// OutputLabeler_8985071583315
// MI455X (gfx1250) — matrix operand order repaired and checked by reading, not compiled, not run on hardware
//
#include <hip/hip_runtime.h>
#include <stdint.h>

// ---- problem constants (from reference) ----
#define B_SZ   8192
#define D_IN   4000
#define EMB    96
#define H0_SZ  2048
#define H1_SZ  1024
#define OPT    32
#define K0_SZ  4096   // D_IN + EMB

typedef __attribute__((ext_vector_type(16))) __bf16        v16bf;
typedef __attribute__((ext_vector_type(8)))  float         v8f;
typedef __attribute__((ext_vector_type(8)))  unsigned int  v8u;

// f32 -> bf16, round-to-nearest-even
__device__ __forceinline__ __bf16 f2bf(float f) {
    unsigned u = __builtin_bit_cast(unsigned, f);
    unsigned r = (u + 0x7FFFu + ((u >> 16) & 1u)) >> 16;
    unsigned short s = (unsigned short)r;
    return __builtin_bit_cast(__bf16, s);
}
__device__ __forceinline__ float bflo(unsigned u) { return __builtin_bit_cast(float, u << 16); }
__device__ __forceinline__ float bfhi(unsigned u) { return __builtin_bit_cast(float, u & 0xFFFF0000u); }

// CDNA5 async fill: global -> LDS, 2 x b128 (32 bytes), tracked by ASYNCcnt.
// INST_OFFSET is added to BOTH the LDS and the global address (ISA 10.x async
// pseudocode), so one base covers both halves.
__device__ __forceinline__ void async_fill_32B(unsigned lds, const __bf16* g) {
    asm volatile("global_load_async_to_lds_b128 %0, %1, off"
                 :: "v"(lds), "v"(g) : "memory");
    asm volatile("global_load_async_to_lds_b128 %0, %1, off offset:16"
                 :: "v"(lds), "v"(g) : "memory");
}

// ---------------------------------------------------------------------------
// Kernel 1: build A = bf16(concat(emb[expert[b]], inputs[b]))  -> [B, 4096]
// ---------------------------------------------------------------------------
__global__ __launch_bounds__(256)
void build_a_kernel(const float* __restrict__ inputs,
                    const float* __restrict__ emb,
                    const int*   __restrict__ ballot,
                    const int*   __restrict__ contest,
                    __bf16*      __restrict__ Abf)
{
    size_t i = (size_t)blockIdx.x * 256 + threadIdx.x;   // B*4096 threads
    int b = (int)(i >> 12);
    int c = (int)(i & 4095);
    int e = ballot[b] * 8 + contest[b];
    float v = (c < EMB) ? emb[e * EMB + c]
                        : inputs[(size_t)b * D_IN + (c - EMB)];
    Abf[i] = f2bf(v);
}

// ---------------------------------------------------------------------------
// Kernel 2: WT[n][k] = bf16(W[k][n])   (tiled transpose + convert)
// ---------------------------------------------------------------------------
__global__ __launch_bounds__(256)
void transpose_conv_kernel(const float* __restrict__ W,
                           __bf16* __restrict__ WT,
                           int K, int N)
{
    __shared__ float tile[32][33];
    const int kb = blockIdx.x * 32;
    const int nb = blockIdx.y * 32;
    const int tx = threadIdx.x & 31;
    const int ty = threadIdx.x >> 5;   // 0..7
#pragma unroll
    for (int j = ty; j < 32; j += 8)
        tile[j][tx] = W[(size_t)(kb + j) * N + (nb + tx)];
    __syncthreads();
#pragma unroll
    for (int j = ty; j < 32; j += 8)
        WT[(size_t)(nb + j) * K + (kb + tx)] = f2bf(tile[tx][j]);
}

// ---------------------------------------------------------------------------
// Kernel 3: C[M,N] = bf16(relu(A[M,K] * BT[N,K]^T + bias[N]))
// Block tile 128x128, 256 threads = 8 waves (4x2), wave tile 32x64.
// Double-buffered LDS filled with global_load_async_to_lds_b128 (ASYNCcnt),
// 8x v_wmma_f32_16x16x32_bf16 per wave per K-step of 32.
// ---------------------------------------------------------------------------
__global__ __launch_bounds__(256, 2)
void gemm_bias_relu_bf16(const __bf16* __restrict__ A,     // [M,K] row-major
                         const __bf16* __restrict__ BT,    // [N,K] row-major
                         const float*  __restrict__ bias,  // [N]
                         __bf16*       __restrict__ C,     // [M,N] row-major
                         int N, int K)
{
    __shared__ __bf16 sA[2][128 * 32];   // [buf][row][k] 2 x 8 KB
    __shared__ __bf16 sB[2][128 * 32];   // [buf][n][k]   2 x 8 KB

    const int tid  = threadIdx.x;
    const int lane = tid & 31;
    const int wave = tid >> 5;
    const int wm   = wave >> 1;       // 0..3 : wave row
    const int wn   = wave & 1;        // 0..1 : wave col
    const int half = lane >> 4;       // 0..1
    const int lan  = lane & 15;

    const int bm = blockIdx.x;
    const int bn = blockIdx.y;

    const v8f vzero = {0.f, 0.f, 0.f, 0.f, 0.f, 0.f, 0.f, 0.f};
    v8f acc[2][4];
#pragma unroll
    for (int mi = 0; mi < 2; ++mi)
#pragma unroll
        for (int ni = 0; ni < 4; ++ni)
            acc[mi][ni] = vzero;

    // tile fill assignment: thread handles 32 bytes of one row (2 threads/row)
    const int r    = tid >> 1;
    const int part = tid & 1;
    const __bf16* gA = A  + (size_t)(bm * 128 + r) * K + part * 16;
    const __bf16* gB = BT + (size_t)(bn * 128 + r) * K + part * 16;
    const unsigned ldsA[2] = { (unsigned)(uintptr_t)&sA[0][r * 32 + part * 16],
                               (unsigned)(uintptr_t)&sA[1][r * 32 + part * 16] };
    const unsigned ldsB[2] = { (unsigned)(uintptr_t)&sB[0][r * 32 + part * 16],
                               (unsigned)(uintptr_t)&sB[1][r * 32 + part * 16] };

    auto issue = [&](int buf, int k0) {
        async_fill_32B(ldsA[buf], gA + k0);   // 2 async b128
        async_fill_32B(ldsB[buf], gB + k0);   // 2 async b128  -> batch = 4
    };

    auto compute = [&](int buf) {
        // A fragment (16x32 bf16): lane half h, VGPR v<4: K=h*8+2v ; v>=4: K=16+h*8+2(v-4)
        v16bf afr[2];
#pragma unroll
        for (int mi = 0; mi < 2; ++mi) {
            const __bf16* rp = &sA[buf][(wm * 32 + mi * 16 + lan) * 32];
            uint4 x = *(const uint4*)(rp + half * 8);        // K = h*8 .. h*8+7
            uint4 y = *(const uint4*)(rp + 16 + half * 8);   // K = 16+h*8 ..
            v8u u = {x.x, x.y, x.z, x.w, y.x, y.y, y.z, y.w};
            afr[mi] = __builtin_bit_cast(v16bf, u);
        }
        // B fragment (32x16 bf16): lane n=lan, VGPR v<4: K=h*8+2v ; v>=4: K=16+h*8+2(v-4)
        v16bf bfr[4];
#pragma unroll
        for (int ni = 0; ni < 4; ++ni) {
            const __bf16* rp = &sB[buf][(wn * 64 + ni * 16 + lan) * 32];
            uint4 x = *(const uint4*)(rp + half * 8);        // K = h*8 .. h*8+7
            uint4 y = *(const uint4*)(rp + 16 + half * 8);   // K = 16+h*8 ..
            v8u u = {x.x, x.y, x.z, x.w, y.x, y.y, y.z, y.w};
            bfr[ni] = __builtin_bit_cast(v16bf, u);
        }
#pragma unroll
        for (int mi = 0; mi < 2; ++mi)
#pragma unroll
            for (int ni = 0; ni < 4; ++ni)
                acc[mi][ni] = __builtin_amdgcn_wmma_f32_16x16x32_bf16(
                    false, afr[mi], false, bfr[ni],
                    (short)0, acc[mi][ni], false, false);
    };

    const int nsteps = K >> 5;
    issue(0, 0);
    for (int i = 0; i < nsteps - 1; ++i) {
        issue((i + 1) & 1, (i + 1) << 5);                    // prefetch next tile
        // async loads complete in order: <=4 outstanding => batch i has landed
        asm volatile("s_wait_asynccnt 0x4" ::: "memory");
        __syncthreads();                                     // whole tile visible
        compute(i & 1);
        __syncthreads();                                     // all reads done before refill
    }
    asm volatile("s_wait_asynccnt 0x0" ::: "memory");
    __syncthreads();
    compute((nsteps - 1) & 1);

    // epilogue: bias + relu, bf16 store. C/D layout: M = h*8+v, N = lan
#pragma unroll
    for (int ni = 0; ni < 4; ++ni) {
        const int n = bn * 128 + wn * 64 + ni * 16 + lan;
        const float bv = bias[n];
#pragma unroll
        for (int mi = 0; mi < 2; ++mi) {
#pragma unroll
            for (int v = 0; v < 8; ++v) {
                const int m = bm * 128 + wm * 32 + mi * 16 + half * 8 + v;
                float x = acc[mi][ni][v] + bv;
                x = x > 0.f ? x : 0.f;
                C[(size_t)m * N + n] = f2bf(x);
            }
        }
    }
}

// ---------------------------------------------------------------------------
// Kernel 4: routed expert head.  One wave per sample, lane = output unit.
// out[b,o] = sigmoid(sum_d h1[b,d]*We[e,d,o] + be[e,o])
// ---------------------------------------------------------------------------
__global__ __launch_bounds__(256)
void routed_out_kernel(const __bf16* __restrict__ H1,   // [B, 1024] bf16
                       const float*  __restrict__ We,   // [64,1024,32]
                       const float*  __restrict__ be,   // [64,32]
                       const int*    __restrict__ ballot,
                       const int*    __restrict__ contest,
                       float*        __restrict__ out)  // [B, 32]
{
    const int b = (int)((blockIdx.x * 256 + threadIdx.x) >> 5);  // sample
    const int o = threadIdx.x & 31;                               // output unit
    const int e = ballot[b] * 8 + contest[b];
    const float* w = We + (size_t)e * H1_SZ * OPT + o;
    const __bf16* h = H1 + (size_t)b * H1_SZ;
    float acc = 0.f;
    for (int d = 0; d < H1_SZ; d += 8) {
        uint4 hu = *(const uint4*)(h + d);     // 8 bf16, lane-uniform address
        acc += bflo(hu.x) * w[(d + 0) * OPT];
        acc += bfhi(hu.x) * w[(d + 1) * OPT];
        acc += bflo(hu.y) * w[(d + 2) * OPT];
        acc += bfhi(hu.y) * w[(d + 3) * OPT];
        acc += bflo(hu.z) * w[(d + 4) * OPT];
        acc += bfhi(hu.z) * w[(d + 5) * OPT];
        acc += bflo(hu.w) * w[(d + 6) * OPT];
        acc += bfhi(hu.w) * w[(d + 7) * OPT];
    }
    acc += be[e * OPT + o];
    out[(size_t)b * OPT + o] = 1.f / (1.f + __expf(-acc));
}

// ---------------------------------------------------------------------------
extern "C" void kernel_launch(void* const* d_in, const int* in_sizes, int n_in,
                              void* d_out, int out_size, void* d_ws, size_t ws_size,
                              hipStream_t stream)
{
    const float* inputs  = (const float*)d_in[0];
    const float* emb     = (const float*)d_in[1];
    const float* W0      = (const float*)d_in[2];
    const float* b0      = (const float*)d_in[3];
    const float* W1      = (const float*)d_in[4];
    const float* b1      = (const float*)d_in[5];
    const float* We      = (const float*)d_in[6];
    const float* be      = (const float*)d_in[7];
    const int*   ballot  = (const int*)d_in[8];
    const int*   contest = (const int*)d_in[9];
    float* out = (float*)d_out;

    // workspace layout (bf16 elements), total ~132 MB
    __bf16* Abf = (__bf16*)d_ws;                         // [8192,4096]  64 MB
    __bf16* W0T = Abf + (size_t)B_SZ * K0_SZ;            // [2048,4096]  16 MB
    __bf16* H0  = W0T + (size_t)H0_SZ * K0_SZ;           // [8192,2048]  32 MB
    __bf16* W1T = H0  + (size_t)B_SZ * H0_SZ;            // [1024,2048]   4 MB
    __bf16* H1  = W1T + (size_t)H1_SZ * H0_SZ;           // [8192,1024]  16 MB

    // 1) gather + concat + bf16 convert
    build_a_kernel<<<(B_SZ * K0_SZ) / 256, 256, 0, stream>>>(
        inputs, emb, ballot, contest, Abf);

    // 2) weight transpose + convert
    transpose_conv_kernel<<<dim3(K0_SZ / 32, H0_SZ / 32), 256, 0, stream>>>(
        W0, W0T, K0_SZ, H0_SZ);
    transpose_conv_kernel<<<dim3(H0_SZ / 32, H1_SZ / 32), 256, 0, stream>>>(
        W1, W1T, H0_SZ, H1_SZ);

    // 3) fc0: [8192,4096] x [4096,2048] -> relu -> bf16
    gemm_bias_relu_bf16<<<dim3(B_SZ / 128, H0_SZ / 128), 256, 0, stream>>>(
        Abf, W0T, b0, H0, H0_SZ, K0_SZ);

    // 4) fc1: [8192,2048] x [2048,1024] -> relu -> bf16
    gemm_bias_relu_bf16<<<dim3(B_SZ / 128, H1_SZ / 128), 256, 0, stream>>>(
        H0, W1T, b1, H1, H1_SZ, H0_SZ);

    // 5) per-sample routed linear + sigmoid
    routed_out_kernel<<<(B_SZ * OPT) / 256, 256, 0, stream>>>(
        H1, We, be, ballot, contest, out);
}
